// FlowNet3D_80556406604131
// MI455X (gfx1250) — compile-verified, matrix operand order checked by reading, not run on hardware
//
#include <hip/hip_runtime.h>
#include <hip/hip_bf16.h>
#include <stdint.h>

typedef _Float16 f16;
typedef _Float16 v16h __attribute__((ext_vector_type(16)));
typedef _Float16 h2   __attribute__((ext_vector_type(2)));
typedef float    v8f  __attribute__((ext_vector_type(8)));

static_assert(sizeof(f16) == 2, "f16 size");

constexpr int BB = 2, N0 = 8192, S1 = 2048, S2 = 512, S3 = 128, S4 = 32;

static inline int pad16i(int x){ return (x + 15) & ~15; }
static inline int pad32i(int x){ return (x + 31) & ~31; }
static inline unsigned cdiv(size_t a, unsigned b){ return (unsigned)((a + (size_t)b - 1) / b); }

// ---------------------------------------------------------------- transpose [B,3,n] -> [B,n,3]
__global__ void k_tr3(const float* __restrict__ in, int n, int total, float* __restrict__ out)
{
    int tid = blockIdx.x * blockDim.x + threadIdx.x;
    if (tid >= total) return;
    int c = tid % 3; int bi = tid / 3; int i = bi % n; int b = bi / n;
    out[tid] = in[((size_t)b * 3 + c) * n + i];
}

// ---------------------------------------------------------------- farthest point sampling
__global__ void k_fps(const float* __restrict__ xyz, int n, int npoint, int* __restrict__ out)
{
    __shared__ float dist[8192];
    __shared__ float rv[256];
    __shared__ int   ri[256];
    int b = blockIdx.x, t = threadIdx.x;
    const float* p = xyz + (size_t)b * n * 3;
    for (int i = t; i < n; i += 256) dist[i] = 1e10f;
    __syncthreads();
    int far = 0;
    for (int it = 0; it < npoint; ++it) {
        if (t == 0) out[b * npoint + it] = far;
        float cx = p[far*3+0], cy = p[far*3+1], cz = p[far*3+2];
        float bestv = -1.0f; int besti = 0;
        for (int i = t; i < n; i += 256) {
            float dx = p[i*3+0]-cx, dy = p[i*3+1]-cy, dz = p[i*3+2]-cz;
            float d = dx*dx + dy*dy + dz*dz;
            float nd = fminf(dist[i], d);
            dist[i] = nd;
            if (nd > bestv) { bestv = nd; besti = i; }
        }
        rv[t] = bestv; ri[t] = besti;
        __syncthreads();
        for (int s = 128; s > 0; s >>= 1) {
            if (t < s) {
                if (rv[t+s] > rv[t] || (rv[t+s] == rv[t] && ri[t+s] < ri[t])) { rv[t] = rv[t+s]; ri[t] = ri[t+s]; }
            }
            __syncthreads();
        }
        far = ri[0];
        __syncthreads();
    }
}

// ---------------------------------------------------------------- gather xyz by fps index
__global__ void k_gather3(const float* __restrict__ in, const int* __restrict__ fidx,
                          int n, int S, int total, float* __restrict__ out)
{
    int tid = blockIdx.x * blockDim.x + threadIdx.x;
    if (tid >= total) return;                 // total = B*S*3
    int c = tid % 3; int bs = tid / 3; int b = bs / S;
    out[tid] = in[((size_t)b * n + fidx[bs]) * 3 + c];
}

// ---------------------------------------------------------------- ball query (first K in-radius, index order, pad first)
__global__ void k_qball(const float* __restrict__ db, int n, const float* __restrict__ q,
                        int S, float r2, int K, int* __restrict__ idx, int total)
{
    int tid = blockIdx.x * blockDim.x + threadIdx.x;
    if (tid >= total) return;                 // total = B*S
    int b = tid / S;
    float qx = q[tid*3+0], qy = q[tid*3+1], qz = q[tid*3+2];
    const float* p = db + (size_t)b * n * 3;
    int* o = idx + (size_t)tid * K;
    int cnt = 0, first = 0; bool got = false;
    for (int i = 0; i < n && cnt < K; ++i) {
        float dx = p[i*3+0]-qx, dy = p[i*3+1]-qy, dz = p[i*3+2]-qz;
        float d = dx*dx + dy*dy + dz*dz;
        if (d <= r2) { if (!got) { got = true; first = i; } o[cnt++] = i; }
    }
    for (; cnt < K; ++cnt) o[cnt] = first;
}

// ---------------------------------------------------------------- kNN (ascending squared distance, stable)
template<int K>
__global__ void k_knn(const float* __restrict__ db, int n, const float* __restrict__ q,
                      int S, int total, int* __restrict__ idx, float* __restrict__ dout)
{
    int tid = blockIdx.x * blockDim.x + threadIdx.x;
    if (tid >= total) return;                 // total = B*S
    int b = tid / S;
    float qx = q[tid*3+0], qy = q[tid*3+1], qz = q[tid*3+2];
    const float* p = db + (size_t)b * n * 3;
    float bd[K]; int bi[K];
    #pragma unroll
    for (int j = 0; j < K; ++j) { bd[j] = 3.4e38f; bi[j] = 0; }
    for (int i = 0; i < n; ++i) {
        float dx = p[i*3+0]-qx, dy = p[i*3+1]-qy, dz = p[i*3+2]-qz;
        float d = dx*dx + dy*dy + dz*dz;
        if (d < bd[K-1]) {
            int j = K - 1;
            while (j > 0 && bd[j-1] > d) { bd[j] = bd[j-1]; bi[j] = bi[j-1]; --j; }
            bd[j] = d; bi[j] = i;
        }
    }
    for (int j = 0; j < K; ++j) {
        idx[(size_t)tid * K + j] = bi[j];
        if (dout) dout[(size_t)tid * K + j] = bd[j];
    }
}

// ---------------------------------------------------------------- pack: set_abstraction group input -> f16 X
__global__ void k_pack_group(const float* __restrict__ dbx, const float* __restrict__ qx,
                             const int* __restrict__ idx, const float* __restrict__ feats,
                             int C, int n, int S, int K, int Kpad, int Mrows, size_t total,
                             f16* __restrict__ X)
{
    size_t e = (size_t)blockIdx.x * blockDim.x + threadIdx.x;
    if (e >= total) return;
    int c = (int)(e % Kpad); size_t row = e / Kpad;
    float v = 0.0f;
    if (row < (size_t)Mrows) {
        int k = (int)(row % K); size_t bs = row / K; int b = (int)(bs / S);
        int j = idx[bs * K + k];
        if (c < 3)            v = dbx[((size_t)b*n + j)*3 + c] - qx[bs*3 + c];
        else if (c < 3 + C)   v = feats[((size_t)b*n + j)*C + (c - 3)];
    }
    X[e] = (f16)v;
}

// ---------------------------------------------------------------- pack: flow embedding input [diff(3), f2g(C), f1(C)]
__global__ void k_pack_fe(const float* __restrict__ dbx, const float* __restrict__ qx,
                          const int* __restrict__ idx, const float* __restrict__ f2,
                          const float* __restrict__ f1, int C, int n, int S, int K,
                          int Kpad, int Mrows, size_t total, f16* __restrict__ X)
{
    size_t e = (size_t)blockIdx.x * blockDim.x + threadIdx.x;
    if (e >= total) return;
    int c = (int)(e % Kpad); size_t row = e / Kpad;
    float v = 0.0f;
    if (row < (size_t)Mrows) {
        int k = (int)(row % K); size_t bs = row / K; int b = (int)(bs / S);
        int j = idx[bs * K + k];
        if (c < 3)              v = dbx[((size_t)b*n + j)*3 + c] - qx[bs*3 + c];
        else if (c < 3 + C)     v = f2[((size_t)b*n + j)*C + (c - 3)];
        else if (c < 3 + 2*C)   v = f1[bs*C + (c - 3 - C)];
    }
    X[e] = (f16)v;
}

// ---------------------------------------------------------------- pack: upconv input [f2g(C2), diff(3)]
__global__ void k_pack_up(const float* __restrict__ dbx, const float* __restrict__ qx,
                          const int* __restrict__ idx, const float* __restrict__ f2,
                          int C2, int n, int S, int K, int Kpad, int Mrows, size_t total,
                          f16* __restrict__ X)
{
    size_t e = (size_t)blockIdx.x * blockDim.x + threadIdx.x;
    if (e >= total) return;
    int c = (int)(e % Kpad); size_t row = e / Kpad;
    float v = 0.0f;
    if (row < (size_t)Mrows) {
        int k = (int)(row % K); size_t bs = row / K; int b = (int)(bs / S);
        int j = idx[bs * K + k];
        if (c < C2)           v = f2[((size_t)b*n + j)*C2 + c];
        else if (c < C2 + 3)  v = dbx[((size_t)b*n + j)*3 + (c - C2)] - qx[bs*3 + (c - C2)];
    }
    X[e] = (f16)v;
}

// ---------------------------------------------------------------- su1: concat+maxpool (no MLP) in f32
__global__ void k_upconv_pool(const float* __restrict__ dbx, const float* __restrict__ qx,
                              const int* __restrict__ idx, const float* __restrict__ f2,
                              int C2, int n, int S, int K, int total, float* __restrict__ out)
{
    int tid = blockIdx.x * blockDim.x + threadIdx.x;
    if (tid >= total) return;                 // total = B*S*(C2+3)
    int Cc = C2 + 3;
    int c = tid % Cc; int bs = tid / Cc; int b = bs / S;
    float m = -3.4e38f;
    for (int k = 0; k < K; ++k) {
        int j = idx[(size_t)bs * K + k];
        float v = (c < C2) ? f2[((size_t)b*n + j)*C2 + c]
                           : dbx[((size_t)b*n + j)*3 + (c - C2)] - qx[(size_t)bs*3 + (c - C2)];
        m = fmaxf(m, v);
    }
    out[(size_t)bs * Cc + c] = m;
}

// ---------------------------------------------------------------- concat up to 3 f32 row-blocks -> f16 X
__global__ void k_concat3(const float* __restrict__ A, int Ca, const float* __restrict__ Bp, int Cb,
                          const float* __restrict__ Cp, int Cc, int Mrows, int Kpad, size_t total,
                          f16* __restrict__ X)
{
    size_t e = (size_t)blockIdx.x * blockDim.x + threadIdx.x;
    if (e >= total) return;
    int c = (int)(e % Kpad); size_t row = e / Kpad;
    float v = 0.0f;
    if (row < (size_t)Mrows) {
        if (c < Ca)                  v = A[row * Ca + c];
        else if (c < Ca + Cb)        v = Bp[row * Cb + (c - Ca)];
        else if (c < Ca + Cb + Cc)   v = Cp[row * Cc + (c - Ca - Cb)];
    }
    X[e] = (f16)v;
}

// ---------------------------------------------------------------- pack: feature propagation [interp(C2), f1t(3)]
__global__ void k_pack_fp(const int* __restrict__ idx, const float* __restrict__ dists,
                          const float* __restrict__ f2, int C2, int n2, int Sq,
                          const float* __restrict__ f1t, int Kpad, int Mrows, size_t total,
                          f16* __restrict__ X)
{
    size_t e = (size_t)blockIdx.x * blockDim.x + threadIdx.x;
    if (e >= total) return;
    int c = (int)(e % Kpad); size_t row = e / Kpad;
    float v = 0.0f;
    if (row < (size_t)Mrows) {
        if (c < C2) {
            int b = (int)(row / Sq);
            float w0 = 1.0f / fmaxf(dists[row*3+0], 1e-10f);
            float w1 = 1.0f / fmaxf(dists[row*3+1], 1e-10f);
            float w2 = 1.0f / fmaxf(dists[row*3+2], 1e-10f);
            float ws = w0 + w1 + w2;
            int j0 = idx[row*3+0], j1 = idx[row*3+1], j2 = idx[row*3+2];
            v = (w0 * f2[((size_t)b*n2 + j0)*C2 + c] +
                 w1 * f2[((size_t)b*n2 + j1)*C2 + c] +
                 w2 * f2[((size_t)b*n2 + j2)*C2 + c]) / ws;
        } else if (c < C2 + 3) {
            v = f1t[row*3 + (c - C2)];
        }
    }
    X[e] = (f16)v;
}

// ---------------------------------------------------------------- weights: W[Cin,Cout] f32 -> Wt[Np,Kpad] f16 (transposed, padded)
__global__ void k_wt(const float* __restrict__ W, int Cin, int Cout, int Kpad, int Np,
                     f16* __restrict__ Wt)
{
    int tid = blockIdx.x * blockDim.x + threadIdx.x;
    if (tid >= Np * Kpad) return;
    int k = tid % Kpad; int co = tid / Kpad;
    float v = 0.0f;
    if (co < Cout && k < Cin) v = W[(size_t)k * Cout + co];
    Wt[tid] = (f16)v;
}

// ---------------------------------------------------------------- WMMA GEMM: Y[M,Cout] = X[Mpad,Kpad] * Wt[Np,Kpad]^T
__global__ __launch_bounds__(256)
void k_gemm(const f16* __restrict__ X, const f16* __restrict__ Wt, float* __restrict__ Y,
            int Kpad, int Cout, int Mrows, int TN, int tiles)
{
    int wave = blockIdx.x * (blockDim.x >> 5) + (threadIdx.x >> 5);
    if (wave >= tiles) return;                       // wave-uniform: EXEC stays all-ones inside
    int lane  = threadIdx.x & 31;
    int half16 = lane >> 4;                          // lane group 0/1
    int l15    = lane & 15;
    int tm = wave / TN, tn = wave % TN;

    const f16* ax = X  + (size_t)(tm * 16 + l15) * Kpad;   // A row (M = l15)
    const f16* bx = Wt + (size_t)(tn * 16 + l15) * Kpad;   // B "row" (N = l15), pre-transposed weights

    v8f acc = {};
    for (int k0 = 0; k0 < Kpad; k0 += 32) {
        if (k0 + 32 < Kpad) __builtin_prefetch((const void*)(ax + k0 + 32), 0, 1);
        v16h av, bv;
        #pragma unroll
        for (int j = 0; j < 8; ++j) {
            // ISA 16-bit A 16x32 layout: VGPR j holds K pair; lanes 16-31 shifted by 8 in K
            int kb = ((j < 4) ? (2 * j) : (16 + 2 * (j - 4))) + (half16 << 3);
            h2 pa = *(const h2*)(ax + k0 + kb);
            h2 pb = *(const h2*)(bx + k0 + kb);
            av[2*j] = pa.x; av[2*j+1] = pa.y;
            bv[2*j] = pb.x; bv[2*j+1] = pb.y;
        }
        acc = __builtin_amdgcn_wmma_f32_16x16x32_f16(false, av, false, bv,
                                                     (short)0, acc, false, false);
    }
    int n = tn * 16 + l15;
    if (n < Cout) {
        #pragma unroll
        for (int r = 0; r < 8; ++r) {                // C/D layout: VGPR r -> M = r + 8*half16
            int m = tm * 16 + r + (half16 << 3);
            if (m < Mrows) Y[(size_t)m * Cout + n] = acc[r];
        }
    }
}

// ---------------------------------------------------------------- BN statistics (mean / rsqrt(var+eps)) per channel
__global__ void k_bn_stats(const float* __restrict__ Y, int Mrows, int C,
                           float* __restrict__ mean, float* __restrict__ rstd)
{
    int c = blockIdx.x, t = threadIdx.x;
    float s = 0.0f, s2 = 0.0f;
    for (int m = t; m < Mrows; m += 256) {
        float v = Y[(size_t)m * C + c];
        s += v; s2 += v * v;
    }
    __shared__ float sh[256], sh2[256];
    sh[t] = s; sh2[t] = s2;
    __syncthreads();
    for (int st = 128; st > 0; st >>= 1) {
        if (t < st) { sh[t] += sh[t+st]; sh2[t] += sh2[t+st]; }
        __syncthreads();
    }
    if (t == 0) {
        float mu = sh[0] / (float)Mrows;
        float var = sh2[0] / (float)Mrows - mu * mu;
        if (var < 0.0f) var = 0.0f;
        mean[c] = mu;
        rstd[c] = rsqrtf(var + 1e-5f);
    }
}

// ---------------------------------------------------------------- BN+ReLU -> padded f16 X for next GEMM
__global__ void k_bn_f16(const float* __restrict__ Y, const float* __restrict__ mean,
                         const float* __restrict__ rstd, int Mrows, int C, int Kpad,
                         size_t total, f16* __restrict__ X)
{
    size_t e = (size_t)blockIdx.x * blockDim.x + threadIdx.x;
    if (e >= total) return;
    int c = (int)(e % Kpad); size_t row = e / Kpad;
    float v = 0.0f;
    if (row < (size_t)Mrows && c < C) {
        v = (Y[row * C + c] - mean[c]) * rstd[c];
        v = v > 0.0f ? v : 0.0f;
    }
    X[e] = (f16)v;
}

// ---------------------------------------------------------------- BN+ReLU -> f32 dst (exact size)
__global__ void k_bn_f32(const float* __restrict__ Y, const float* __restrict__ mean,
                         const float* __restrict__ rstd, int Mrows, int C,
                         float* __restrict__ dst, size_t total)
{
    size_t e = (size_t)blockIdx.x * blockDim.x + threadIdx.x;
    if (e >= total) return;
    int c = (int)(e % C);
    float v = (Y[e] - mean[c]) * rstd[c];
    dst[e] = v > 0.0f ? v : 0.0f;
}

// ---------------------------------------------------------------- max pool over K grouped rows
__global__ void k_maxpool(const float* __restrict__ src, int K, int C, int rowsOut,
                          float* __restrict__ dst, size_t total)
{
    size_t e = (size_t)blockIdx.x * blockDim.x + threadIdx.x;
    if (e >= total) return;
    int c = (int)(e % C); size_t r = e / C;
    float m = -3.4e38f;
    for (int k = 0; k < K; ++k) m = fmaxf(m, src[((size_t)r * K + k) * C + c]);
    dst[e] = m;
}

// ---------------------------------------------------------------- head: add bias + transpose [B,N,3]->[B,3,N]
__global__ void k_head_out(const float* __restrict__ Y, const float* __restrict__ bias,
                           float* __restrict__ out, int n, int total)
{
    int tid = blockIdx.x * blockDim.x + threadIdx.x;
    if (tid >= total) return;
    int i = tid % n; int bc = tid / n; int c = bc % 3; int b = bc / 3;
    out[tid] = Y[((size_t)b * n + i) * 3 + c] + bias[c];
}

// ================================================================ host orchestration
namespace {

struct Bump {
    uint8_t* p; size_t off;
    void* take(size_t bytes) { off = (off + 255) & ~(size_t)255; void* r = p + off; off += bytes; return r; }
};

struct Ctx {
    hipStream_t st;
    f16 *Xa, *Xb, *Wt;
    float *Y, *mean, *rstd;
};

void conv_bn(const Ctx& c, const f16* X, int Kpad, const float* W, int Cin, int Cout, int Mrows)
{
    int Np = pad16i(Cout);
    k_wt<<<cdiv((size_t)Np * Kpad, 256), 256, 0, c.st>>>(W, Cin, Cout, Kpad, Np, c.Wt);
    int TN = Np / 16;
    int tiles = (Mrows / 16) * TN;
    k_gemm<<<(tiles + 7) / 8, 256, 0, c.st>>>(X, c.Wt, c.Y, Kpad, Cout, Mrows, TN, tiles);
    k_bn_stats<<<Cout, 256, 0, c.st>>>(c.Y, Mrows, Cout, c.mean, c.rstd);
}

void fin_f16(const Ctx& c, int Mrows, int Cout, int KpadN, f16* Xn)
{
    size_t tot = (size_t)Mrows * KpadN;
    k_bn_f16<<<cdiv(tot, 256), 256, 0, c.st>>>(c.Y, c.mean, c.rstd, Mrows, Cout, KpadN, tot, Xn);
}

void fin_f32(const Ctx& c, int Mrows, int Cout, float* dst)
{
    size_t tot = (size_t)Mrows * Cout;
    k_bn_f32<<<cdiv(tot, 256), 256, 0, c.st>>>(c.Y, c.mean, c.rstd, Mrows, Cout, dst, tot);
}

// MLP chain starting from X in c.Xa (pre-packed with Kpad = pad32(Cin)).
// poolK > 0: final BN+ReLU in-place on Y, then max over poolK -> dst [Mrows/poolK, Clast]
// poolK == 0: final BN+ReLU -> dst [Mrows, Clast]
void mlp_chain(const Ctx& c, const float* const* Ws, const int* co, int nl,
               int Cin, int Mrows, int poolK, float* dst)
{
    const f16* X = c.Xa;
    int Kp = pad32i(Cin), cin = Cin;
    for (int i = 0; i < nl; ++i) {
        conv_bn(c, X, Kp, Ws[i], cin, co[i], Mrows);
        if (i + 1 < nl) {
            f16* Xn = (X == c.Xa) ? c.Xb : c.Xa;
            int Kn = pad32i(co[i]);
            fin_f16(c, Mrows, co[i], Kn, Xn);
            X = Xn; Kp = Kn; cin = co[i];
        }
    }
    int Cl = co[nl - 1];
    if (poolK > 0) {
        fin_f32(c, Mrows, Cl, c.Y);     // in-place relu/normalize
        int rowsOut = Mrows / poolK;
        size_t tot = (size_t)rowsOut * Cl;
        k_maxpool<<<cdiv(tot, 256), 256, 0, c.st>>>(c.Y, poolK, Cl, rowsOut, dst, tot);
    } else {
        fin_f32(c, Mrows, Cl, dst);
    }
}

} // namespace

extern "C" void kernel_launch(void* const* d_in, const int* in_sizes, int n_in,
                              void* d_out, int out_size, void* d_ws, size_t ws_size,
                              hipStream_t stream)
{
    (void)in_sizes; (void)n_in; (void)out_size; (void)ws_size;

    // ---- input map (setup_inputs dict order, params flattened) ----
    auto Wp = [&](int i) { return (const float*)d_in[i]; };
    const float* pc[2] = { (const float*)d_in[0], (const float*)d_in[1] };
    const float* ft[2] = { (const float*)d_in[2], (const float*)d_in[3] };
    enum { W_SA1 = 4, W_SA2 = 7, W_SA3 = 10, W_SA4 = 13, W_FE = 16,
           W_SU1_2 = 19, W_SU2_1 = 21, W_SU2_2 = 24, W_SU3_1 = 25, W_SU3_2 = 28,
           W_FP = 29, W_CONV1 = 31, W_CONV2W = 32, W_CONV2B = 33 };

    // ---- workspace layout (deterministic bump) ----
    Bump ws{ (uint8_t*)d_ws, 0 };
    auto F = [&](size_t n) { return (float*)ws.take(n * sizeof(float)); };
    auto I = [&](size_t n) { return (int*)ws.take(n * sizeof(int)); };
    auto H = [&](size_t n) { return (f16*)ws.take(n * sizeof(f16)); };

    float *xyz0[2], *f0t[2], *l1p[2], *l2p[2], *l1f[2], *l2f[2];
    for (int cl = 0; cl < 2; ++cl) {
        xyz0[cl] = F((size_t)BB * N0 * 3);
        f0t[cl]  = F((size_t)BB * N0 * 3);
        l1p[cl]  = F((size_t)BB * S1 * 3);
        l2p[cl]  = F((size_t)BB * S2 * 3);
        l1f[cl]  = F((size_t)BB * S1 * 64);
        l2f[cl]  = F((size_t)BB * S2 * 128);
    }
    float* l3p   = F((size_t)BB * S3 * 3);
    float* l4p   = F((size_t)BB * S4 * 3);
    float* l3f   = F((size_t)BB * S3 * 256);
    float* l4f   = F((size_t)BB * S4 * 512);
    float* l2f1n = F((size_t)BB * S2 * 128);
    float* l3fn  = F((size_t)BB * S3 * 256);
    float* l2fn  = F((size_t)BB * S2 * 256);
    float* l1fn  = F((size_t)BB * S1 * 256);
    float* pooled= F((size_t)BB * S1 * 256);          // covers all upconv pooled temps
    float* dists = F((size_t)BB * N0 * 3);
    int*   fidx  = I((size_t)BB * S1);
    int*   gidx  = I(65536);

    Ctx c;
    c.st = stream;
    c.Xa = H((size_t)65536 * 288);                    // biggest packed X (flow embedding)
    c.Xb = H((size_t)65536 * 128);                    // biggest intermediate X
    c.Wt = H((size_t)256 * 832);                      // biggest transposed weight (su1_2 L0)
    c.Y  = F((size_t)65536 * 128);                    // biggest GEMM output
    float* stats = F(1024);
    c.mean = stats; c.rstd = stats + 512;

    // ================= per-cloud: transpose, sa1, sa2 =================
    for (int cl = 0; cl < 2; ++cl) {
        int tot3 = BB * N0 * 3;
        k_tr3<<<cdiv((size_t)tot3, 256), 256, 0, stream>>>(pc[cl], N0, tot3, xyz0[cl]);
        k_tr3<<<cdiv((size_t)tot3, 256), 256, 0, stream>>>(ft[cl], N0, tot3, f0t[cl]);

        // ---- sa1: 8192 -> 2048, r=5, K=16, MLP 6->32->32->64 ----
        k_fps<<<BB, 256, 0, stream>>>(xyz0[cl], N0, S1, fidx);
        k_gather3<<<cdiv((size_t)BB*S1*3, 256), 256, 0, stream>>>(xyz0[cl], fidx, N0, S1, BB*S1*3, l1p[cl]);
        k_qball<<<cdiv((size_t)BB*S1, 256), 256, 0, stream>>>(xyz0[cl], N0, l1p[cl], S1, 25.0f, 16, gidx, BB*S1);
        {
            int Mr = BB * S1 * 16, Kp = pad32i(6);
            size_t tot = (size_t)Mr * Kp;
            k_pack_group<<<cdiv(tot, 256), 256, 0, stream>>>(xyz0[cl], l1p[cl], gidx, f0t[cl],
                                                             3, N0, S1, 16, Kp, Mr, tot, c.Xa);
            const float* Wl[3] = { Wp(W_SA1), Wp(W_SA1+1), Wp(W_SA1+2) };
            int co[3] = { 32, 32, 64 };
            mlp_chain(c, Wl, co, 3, 6, Mr, 16, l1f[cl]);
        }

        // ---- sa2: 2048 -> 512, r=10, K=16, MLP 67->64->64->128 ----
        k_fps<<<BB, 256, 0, stream>>>(l1p[cl], S1, S2, fidx);
        k_gather3<<<cdiv((size_t)BB*S2*3, 256), 256, 0, stream>>>(l1p[cl], fidx, S1, S2, BB*S2*3, l2p[cl]);
        k_qball<<<cdiv((size_t)BB*S2, 256), 256, 0, stream>>>(l1p[cl], S1, l2p[cl], S2, 100.0f, 16, gidx, BB*S2);
        {
            int Mr = BB * S2 * 16, Kp = pad32i(67);
            size_t tot = (size_t)Mr * Kp;
            k_pack_group<<<cdiv(tot, 256), 256, 0, stream>>>(l1p[cl], l2p[cl], gidx, l1f[cl],
                                                             64, S1, S2, 16, Kp, Mr, tot, c.Xa);
            const float* Wl[3] = { Wp(W_SA2), Wp(W_SA2+1), Wp(W_SA2+2) };
            int co[3] = { 64, 64, 128 };
            mlp_chain(c, Wl, co, 3, 67, Mr, 16, l2f[cl]);
        }
    }

    // ================= flow embedding: knn64 on cloud2, MLP 259->128->128->128 =================
    k_knn<64><<<cdiv((size_t)BB*S2, 256), 256, 0, stream>>>(l2p[1], S2, l2p[0], S2, BB*S2, gidx, nullptr);
    {
        int Mr = BB * S2 * 64, Kp = pad32i(259);
        size_t tot = (size_t)Mr * Kp;
        k_pack_fe<<<cdiv(tot, 256), 256, 0, stream>>>(l2p[1], l2p[0], gidx, l2f[1], l2f[0],
                                                      128, S2, S2, 64, Kp, Mr, tot, c.Xa);
        const float* Wl[3] = { Wp(W_FE), Wp(W_FE+1), Wp(W_FE+2) };
        int co[3] = { 128, 128, 128 };
        mlp_chain(c, Wl, co, 3, 259, Mr, 64, l2f1n);
    }

    // ================= sa3: 512 -> 128, r=17.5, K=8, MLP 131->128->128->256 =================
    k_fps<<<BB, 256, 0, stream>>>(l2p[0], S2, S3, fidx);
    k_gather3<<<cdiv((size_t)BB*S3*3, 256), 256, 0, stream>>>(l2p[0], fidx, S2, S3, BB*S3*3, l3p);
    k_qball<<<cdiv((size_t)BB*S3, 256), 256, 0, stream>>>(l2p[0], S2, l3p, S3, 306.25f, 8, gidx, BB*S3);
    {
        int Mr = BB * S3 * 8, Kp = pad32i(131);
        size_t tot = (size_t)Mr * Kp;
        k_pack_group<<<cdiv(tot, 256), 256, 0, stream>>>(l2p[0], l3p, gidx, l2f1n,
                                                         128, S2, S3, 8, Kp, Mr, tot, c.Xa);
        const float* Wl[3] = { Wp(W_SA3), Wp(W_SA3+1), Wp(W_SA3+2) };
        int co[3] = { 128, 128, 256 };
        mlp_chain(c, Wl, co, 3, 131, Mr, 8, l3f);
    }

    // ================= sa4: 128 -> 32, r=25, K=8, MLP 259->256->256->512 =================
    k_fps<<<BB, 256, 0, stream>>>(l3p, S3, S4, fidx);
    k_gather3<<<cdiv((size_t)BB*S4*3, 256), 256, 0, stream>>>(l3p, fidx, S3, S4, BB*S4*3, l4p);
    k_qball<<<cdiv((size_t)BB*S4, 256), 256, 0, stream>>>(l3p, S3, l4p, S4, 625.0f, 8, gidx, BB*S4);
    {
        int Mr = BB * S4 * 8, Kp = pad32i(259);
        size_t tot = (size_t)Mr * Kp;
        k_pack_group<<<cdiv(tot, 256), 256, 0, stream>>>(l3p, l4p, gidx, l3f,
                                                         256, S3, S4, 8, Kp, Mr, tot, c.Xa);
        const float* Wl[3] = { Wp(W_SA4), Wp(W_SA4+1), Wp(W_SA4+2) };
        int co[3] = { 256, 256, 512 };
        mlp_chain(c, Wl, co, 3, 259, Mr, 8, l4f);
    }

    // ================= su1: l4 -> l3 (no pre-MLP), concat pooled(515)+l3f(256), MLP 771->256->256 =================
    k_knn<8><<<cdiv((size_t)BB*S3, 256), 256, 0, stream>>>(l4p, S4, l3p, S3, BB*S3, gidx, nullptr);
    {
        int tot = BB * S3 * (512 + 3);
        k_upconv_pool<<<cdiv((size_t)tot, 256), 256, 0, stream>>>(l4p, l3p, gidx, l4f,
                                                                  512, S4, S3, 8, tot, pooled);
        int Mr = BB * S3, Kp = pad32i(771);
        size_t ctot = (size_t)Mr * Kp;
        k_concat3<<<cdiv(ctot, 256), 256, 0, stream>>>(pooled, 515, l3f, 256, nullptr, 0, Mr, Kp, ctot, c.Xa);
        const float* Wl[2] = { Wp(W_SU1_2), Wp(W_SU1_2+1) };
        int co[2] = { 256, 256 };
        mlp_chain(c, Wl, co, 2, 771, Mr, 0, l3fn);
    }

    // ================= su2: l3 -> l2, MLP1 259->128->128->256 + pool, concat(256+128+128), MLP2 512->256 =================
    k_knn<8><<<cdiv((size_t)BB*S2, 256), 256, 0, stream>>>(l3p, S3, l2p[0], S2, BB*S2, gidx, nullptr);
    {
        int Mr = BB * S2 * 8, Kp = pad32i(259);
        size_t tot = (size_t)Mr * Kp;
        k_pack_up<<<cdiv(tot, 256), 256, 0, stream>>>(l3p, l2p[0], gidx, l3fn,
                                                      256, S3, S2, 8, Kp, Mr, tot, c.Xa);
        const float* W1[3] = { Wp(W_SU2_1), Wp(W_SU2_1+1), Wp(W_SU2_1+2) };
        int co1[3] = { 128, 128, 256 };
        mlp_chain(c, W1, co1, 3, 259, Mr, 8, pooled);

        int Mr2 = BB * S2, Kp2 = pad32i(512);
        size_t ctot = (size_t)Mr2 * Kp2;
        k_concat3<<<cdiv(ctot, 256), 256, 0, stream>>>(pooled, 256, l2f[0], 128, l2f1n, 128, Mr2, Kp2, ctot, c.Xa);
        const float* W2[1] = { Wp(W_SU2_2) };
        int co2[1] = { 256 };
        mlp_chain(c, W2, co2, 1, 512, Mr2, 0, l2fn);
    }

    // ================= su3: l2 -> l1, MLP1 259->128->128->256 + pool, concat(256+64), MLP2 320->256 =================
    k_knn<8><<<cdiv((size_t)BB*S1, 256), 256, 0, stream>>>(l2p[0], S2, l1p[0], S1, BB*S1, gidx, nullptr);
    {
        int Mr = BB * S1 * 8, Kp = pad32i(259);
        size_t tot = (size_t)Mr * Kp;
        k_pack_up<<<cdiv(tot, 256), 256, 0, stream>>>(l2p[0], l1p[0], gidx, l2fn,
                                                      256, S2, S1, 8, Kp, Mr, tot, c.Xa);
        const float* W1[3] = { Wp(W_SU3_1), Wp(W_SU3_1+1), Wp(W_SU3_1+2) };
        int co1[3] = { 128, 128, 256 };
        mlp_chain(c, W1, co1, 3, 259, Mr, 8, pooled);

        int Mr2 = BB * S1, Kp2 = pad32i(320);
        size_t ctot = (size_t)Mr2 * Kp2;
        k_concat3<<<cdiv(ctot, 256), 256, 0, stream>>>(pooled, 256, l1f[0], 64, nullptr, 0, Mr2, Kp2, ctot, c.Xa);
        const float* W2[1] = { Wp(W_SU3_2) };
        int co2[1] = { 256 };
        mlp_chain(c, W2, co2, 1, 320, Mr2, 0, l1fn);
    }

    // ================= feature prop + head (fp0, fp1, conv1 all conv+bn+relu; then conv2+bias) =================
    k_knn<3><<<cdiv((size_t)BB*N0, 256), 256, 0, stream>>>(l1p[0], S1, xyz0[0], N0, BB*N0, gidx, dists);
    {
        int Mr = BB * N0, Kp = pad32i(259);
        size_t tot = (size_t)Mr * Kp;
        k_pack_fp<<<cdiv(tot, 256), 256, 0, stream>>>(gidx, dists, l1fn, 256, S1, N0, f0t[0],
                                                      Kp, Mr, tot, c.Xa);
        conv_bn(c, c.Xa, Kp,  Wp(W_FP),    259, 256, Mr);  fin_f16(c, Mr, 256, 256, c.Xb);
        conv_bn(c, c.Xb, 256, Wp(W_FP+1),  256, 256, Mr);  fin_f16(c, Mr, 256, 256, c.Xa);
        conv_bn(c, c.Xa, 256, Wp(W_CONV1), 256, 128, Mr);  fin_f16(c, Mr, 128, 128, c.Xb);

        // conv2 (no BN): GEMM 128 -> 3 (padded to 16 cols), then bias + transpose out
        int Np = 16;
        k_wt<<<cdiv((size_t)Np * 128, 256), 256, 0, stream>>>(Wp(W_CONV2W), 128, 3, 128, Np, c.Wt);
        int tiles = (Mr / 16) * (Np / 16);
        k_gemm<<<(tiles + 7) / 8, 256, 0, stream>>>(c.Xb, c.Wt, c.Y, 128, 3, Mr, Np / 16, tiles);

        int otot = BB * 3 * N0;
        k_head_out<<<cdiv((size_t)otot, 256), 256, 0, stream>>>(c.Y, Wp(W_CONV2B), (float*)d_out, N0, otot);
    }
}
